// GPT2RoPEAttention_88691074662495
// MI455X (gfx1250) — matrix operand order repaired and checked by reading, not compiled, not run on hardware
//
#include <hip/hip_runtime.h>

typedef __attribute__((ext_vector_type(16))) _Float16 v16h;
typedef __attribute__((ext_vector_type(8)))  float    v8f;

#define WMMA_F16(A, B, C) \
    __builtin_amdgcn_wmma_f32_16x16x32_f16(false, (A), false, (B), (short)0, (C), false, false)

constexpr int Bc  = 2;
constexpr int Sc  = 2048;
constexpr int Ec  = 1024;
constexpr int Hc  = 16;
constexpr int Dc  = 64;
constexpr int N3c = 3 * Ec;

// ---------------- fragment loaders (layouts per CDNA5 ISA 7.12.2) ----------------

// A fragment 16x32 (f16) from row-major f32 src.
// lane<16: row=lane, K = {kb..kb+7, kb+16..kb+23}; lane>=16: row=lane-16, K +8.
__device__ __forceinline__ v16h load_a_f32(const float* __restrict__ A, int lda,
                                           int mb, int kb, int lane) {
    const float* p = A + (size_t)(mb + (lane & 15)) * lda + kb + ((lane >> 4) << 3);
    v16h a;
#pragma unroll
    for (int i = 0; i < 8; ++i) { a[i] = (_Float16)p[i]; a[8 + i] = (_Float16)p[i + 16]; }
    return a;
}

__device__ __forceinline__ v16h load_a_f16(const _Float16* __restrict__ A, int lda,
                                           int mb, int kb, int lane) {
    const _Float16* p = A + (size_t)(mb + (lane & 15)) * lda + kb + ((lane >> 4) << 3);
    v16h a;
#pragma unroll
    for (int i = 0; i < 8; ++i) { a[i] = p[i]; a[8 + i] = p[i + 16]; }
    return a;
}

// B fragment 32x16 where B = M^T and M (N x K) is row-major: two runs of 8 halves/lane.
// B[k][n] = M[n][k]; lane: n = nb + lane%16, K = {kb..kb+7, kb+16..kb+23} (+8 for lane>=16).
__device__ __forceinline__ v16h load_bt_f16(const _Float16* __restrict__ M, int ldm,
                                            int nb, int kb, int lane) {
    const _Float16* p = M + (size_t)(nb + (lane & 15)) * ldm + kb + ((lane >> 4) << 3);
    v16h b;
#pragma unroll
    for (int i = 0; i < 8; ++i) { b[i] = p[i]; b[8 + i] = p[i + 16]; }
    return b;
}

// reductions across the 16-lane half (rows of a C fragment span lanes 0-15 / 16-31)
__device__ __forceinline__ float half_max(float v) {
#pragma unroll
    for (int off = 1; off < 16; off <<= 1) v = fmaxf(v, __shfl_xor(v, off, 32));
    return v;
}
__device__ __forceinline__ float half_sum(float v) {
#pragma unroll
    for (int off = 1; off < 16; off <<= 1) v += __shfl_xor(v, off, 32);
    return v;
}

// ---------------- kernel 0: transpose + f32->f16 convert of a weight matrix --------
// src: K x N row-major f32; dst: N x K row-major f16. block (32,8), grid (N/32, K/32).
__global__ void transpose_w_kernel(const float* __restrict__ src, _Float16* __restrict__ dst,
                                   int K, int N) {
    __shared__ float tile[32][33];
    const int tx = threadIdx.x, ty = threadIdx.y;
    const int n0 = blockIdx.x << 5, k0 = blockIdx.y << 5;
#pragma unroll
    for (int i = 0; i < 4; ++i)
        tile[ty + 8 * i][tx] = src[(size_t)(k0 + ty + 8 * i) * N + n0 + tx];
    __syncthreads();
#pragma unroll
    for (int i = 0; i < 4; ++i)
        dst[(size_t)(n0 + ty + 8 * i) * K + k0 + tx] = (_Float16)tile[tx][ty + 8 * i];
}

// ---------------- kernel 1: QKV GEMM + bias + RoPE, f16 pack ----------------
// grid (N3c/64, (Bc*Sc)/32), block 32 (one wave, 32x64 tile, 8 wmma / k-step).
// qkv layout: chunk 0/1 (Q,K): (B,H,S,D); chunk 2 (V): transposed (B,H,D,S).
__global__ void qkv_rope_kernel(const float* __restrict__ hs, const _Float16* __restrict__ Wt,
                                const float* __restrict__ bias, _Float16* __restrict__ qkv) {
    const int lane = threadIdx.x & 31;
    const int mb = blockIdx.y << 5;
    const int nb = blockIdx.x << 6;
    v8f acc[2][4] = {};
    for (int k = 0; k < Ec; k += 32) {
        if (k + 32 < Ec) {   // L2-latency hiding: global_prefetch_b8
            __builtin_prefetch(hs + (size_t)(mb + (lane & 15)) * Ec + k + 32, 0, 1);
            __builtin_prefetch(Wt + (size_t)(nb + (lane & 15)) * Ec + k + 32, 0, 1);
        }
        const v16h a0 = load_a_f32(hs, Ec, mb, k, lane);
        const v16h a1 = load_a_f32(hs, Ec, mb + 16, k, lane);
        v16h bf[4];
#pragma unroll
        for (int ni = 0; ni < 4; ++ni) bf[ni] = load_bt_f16(Wt, Ec, nb + ni * 16, k, lane);
#pragma unroll
        for (int ni = 0; ni < 4; ++ni) {
            acc[0][ni] = WMMA_F16(a0, bf[ni], acc[0][ni]);
            acc[1][ni] = WMMA_F16(a1, bf[ni], acc[1][ni]);
        }
    }
    const int half = lane >> 4, col = lane & 15;
    const size_t plane = (size_t)Sc * Dc;
#pragma unroll
    for (int mi = 0; mi < 2; ++mi) {
#pragma unroll
        for (int ni = 0; ni < 4; ++ni) {
#pragma unroll
            for (int j = 0; j < 8; ++j) {
                const int m = mb + mi * 16 + j + (half << 3);
                const int n = nb + ni * 16 + col;
                float v = acc[mi][ni][j] + bias[n];
                // RoPE pair partner lives in lane^1 (d parity == lane parity)
                const float pv = __shfl_xor(v, 1, 32);
                const int chunk = n >> 10, e = n & 1023, h = e >> 6, d = e & 63;
                float r = v;
                if (chunk < 2) {  // q,k only; reference uses HEAD index as position
                    const float freq = __expf(-0.14391157f * (float)(d & ~1)); // ln(1e4)/64
                    const float ang  = (float)h * freq;
                    float sn, cs;
                    __sincosf(ang, &sn, &cs);
                    r = (d & 1) ? (pv * sn + v * cs) : (v * cs - pv * sn);
                }
                const int bb = m >> 11, si = m & 2047;
                const size_t base = (((size_t)chunk * Bc + bb) * Hc + h) * plane;
                const size_t off  = (chunk == 2) ? ((size_t)d * Sc + si)   // V transposed
                                                 : ((size_t)si * Dc + d);  // Q,K row-major
                qkv[base + off] = (_Float16)r;
            }
        }
    }
}

// ---------------- kernel 2: causal flash attention ----------------
// grid (Sc/16, Bc*Hc), block 32 (one wave per 16-query tile). obuf: (B,S,E) f16.
__global__ void flash_attn_kernel(const _Float16* __restrict__ qkv,
                                  _Float16* __restrict__ obuf) {
    __shared__ _Float16 pbuf[16 * 32];  // P tile bounce: C-layout -> A-layout
    const int lane = threadIdx.x & 31;
    const int half = lane >> 4, col = lane & 15;
    const int qt = blockIdx.x;
    const int b  = blockIdx.y / Hc, h = blockIdx.y % Hc;
    const size_t plane = (size_t)Sc * Dc;
    const _Float16* Q  = qkv + ((size_t)(0 * Bc + b) * Hc + h) * plane;  // (S,D)
    const _Float16* K  = qkv + ((size_t)(1 * Bc + b) * Hc + h) * plane;  // (S,D)
    const _Float16* Vt = qkv + ((size_t)(2 * Bc + b) * Hc + h) * plane;  // (D,S)

    const v16h aq0 = load_a_f16(Q, Dc, qt * 16, 0, lane);
    const v16h aq1 = load_a_f16(Q, Dc, qt * 16, 32, lane);

    float mrow[8], lrow[8];
#pragma unroll
    for (int j = 0; j < 8; ++j) { mrow[j] = -1e30f; lrow[j] = 0.0f; }
    v8f o0 = {}, o1 = {}, o2 = {}, o3 = {};

    const int qlast = qt * 16 + 15;
    for (int kb = 0; kb <= qlast; kb += 32) {
        if (kb + 32 <= qlast) {  // prefetch next key tile (K rows + Vt rows)
            __builtin_prefetch(K + (size_t)(kb + 32 + col) * Dc, 0, 1);
            __builtin_prefetch(Vt + (size_t)col * Sc + kb + 32, 0, 1);
        }
        // scores: Q(16x64) x K^T(64x32) -> two 16x16 C frags (contiguous K-row loads)
        const v16h b00 = load_bt_f16(K, Dc, kb, 0, lane);
        const v16h b01 = load_bt_f16(K, Dc, kb, 32, lane);
        const v16h b10 = load_bt_f16(K, Dc, kb + 16, 0, lane);
        const v16h b11 = load_bt_f16(K, Dc, kb + 16, 32, lane);
        v8f s0 = {}, s1 = {};
        s0 = WMMA_F16(aq0, b00, s0);
        s0 = WMMA_F16(aq1, b01, s0);
        s1 = WMMA_F16(aq0, b10, s1);
        s1 = WMMA_F16(aq1, b11, s1);

        const bool need_mask = (kb + 31 > qt * 16);
        float p0[8], p1[8];
#pragma unroll
        for (int j = 0; j < 8; ++j) {
            const int row = qt * 16 + j + (half << 3);
            float x0 = s0[j] * 0.125f;   // 1/sqrt(64)
            float x1 = s1[j] * 0.125f;
            if (need_mask) {
                if (kb + col > row)      x0 = -1e30f;
                if (kb + 16 + col > row) x1 = -1e30f;
            }
            const float tm    = half_max(fmaxf(x0, x1));
            const float mnew  = fmaxf(mrow[j], tm);
            const float alpha = __expf(mrow[j] - mnew);
            x0 = __expf(x0 - mnew);
            x1 = __expf(x1 - mnew);
            const float rs = half_sum(x0 + x1);
            lrow[j] = lrow[j] * alpha + rs;
            mrow[j] = mnew;
            o0[j] *= alpha; o1[j] *= alpha; o2[j] *= alpha; o3[j] *= alpha;
            p0[j] = x0; p1[j] = x1;
        }

        // P: C-layout -> LDS (row-major 16x32) -> A-fragment
        __syncthreads();
#pragma unroll
        for (int j = 0; j < 8; ++j) {
            const int r = j + (half << 3);
            pbuf[r * 32 + col]      = (_Float16)p0[j];
            pbuf[r * 32 + 16 + col] = (_Float16)p1[j];
        }
        __syncthreads();
        const v16h ap = load_a_f16(pbuf, 32, 0, 0, lane);
        // P(16x32) x V(32x64): V^T rows are contiguous over keys
        o0 = WMMA_F16(ap, load_bt_f16(Vt, Sc,  0, kb, lane), o0);
        o1 = WMMA_F16(ap, load_bt_f16(Vt, Sc, 16, kb, lane), o1);
        o2 = WMMA_F16(ap, load_bt_f16(Vt, Sc, 32, kb, lane), o2);
        o3 = WMMA_F16(ap, load_bt_f16(Vt, Sc, 48, kb, lane), o3);
    }

#pragma unroll
    for (int j = 0; j < 8; ++j) {
        const int s = qt * 16 + j + (half << 3);
        const float inv = 1.0f / lrow[j];
        const size_t base = ((size_t)b * Sc + s) * Ec + (size_t)h * Dc + col;
        obuf[base +  0] = (_Float16)(o0[j] * inv);
        obuf[base + 16] = (_Float16)(o1[j] * inv);
        obuf[base + 32] = (_Float16)(o2[j] * inv);
        obuf[base + 48] = (_Float16)(o3[j] * inv);
    }
}

// ---------------- kernel 3: output projection ----------------
// grid (Ec/64, (Bc*Sc)/32), block 32 (32x64 tile, 8 wmma / k-step).
__global__ void proj_kernel(const _Float16* __restrict__ A, const _Float16* __restrict__ Wt,
                            const float* __restrict__ bias, float* __restrict__ out) {
    const int lane = threadIdx.x & 31;
    const int mb = blockIdx.y << 5, nb = blockIdx.x << 6;
    v8f acc[2][4] = {};
    for (int k = 0; k < Ec; k += 32) {
        if (k + 32 < Ec) {
            __builtin_prefetch(A + (size_t)(mb + (lane & 15)) * Ec + k + 32, 0, 1);
            __builtin_prefetch(Wt + (size_t)(nb + (lane & 15)) * Ec + k + 32, 0, 1);
        }
        const v16h a0 = load_a_f16(A, Ec, mb, k, lane);
        const v16h a1 = load_a_f16(A, Ec, mb + 16, k, lane);
        v16h bf[4];
#pragma unroll
        for (int ni = 0; ni < 4; ++ni) bf[ni] = load_bt_f16(Wt, Ec, nb + ni * 16, k, lane);
#pragma unroll
        for (int ni = 0; ni < 4; ++ni) {
            acc[0][ni] = WMMA_F16(a0, bf[ni], acc[0][ni]);
            acc[1][ni] = WMMA_F16(a1, bf[ni], acc[1][ni]);
        }
    }
    const int half = lane >> 4, col = lane & 15;
#pragma unroll
    for (int ni = 0; ni < 4; ++ni) {
#pragma unroll
        for (int j = 0; j < 8; ++j) {
            const int m0 = mb + j + (half << 3);
            const int n  = nb + ni * 16 + col;
            out[(size_t)m0 * Ec + n]        = acc[0][ni][j] + bias[n];
            out[(size_t)(m0 + 16) * Ec + n] = acc[1][ni][j] + bias[n];
        }
    }
}

// ---------------- launcher ----------------
extern "C" void kernel_launch(void* const* d_in, const int* in_sizes, int n_in,
                              void* d_out, int out_size, void* d_ws, size_t ws_size,
                              hipStream_t stream) {
    (void)in_sizes; (void)n_in; (void)out_size; (void)ws_size;
    const float* hs = (const float*)d_in[0];
    const float* Wa = (const float*)d_in[1];
    const float* ba = (const float*)d_in[2];
    const float* Wp = (const float*)d_in[3];
    const float* bp = (const float*)d_in[4];
    float* out = (float*)d_out;

    char* ws = (char*)d_ws;
    const size_t qkv_bytes  = (size_t)3 * Bc * Hc * Sc * Dc * sizeof(_Float16); // 24 MB
    const size_t obuf_bytes = (size_t)Bc * Sc * Ec * sizeof(_Float16);          //  8 MB
    const size_t wat_bytes  = (size_t)N3c * Ec * sizeof(_Float16);              //  6 MB
    _Float16* qkv  = (_Float16*)ws;
    _Float16* obuf = (_Float16*)(ws + qkv_bytes);
    _Float16* WaT  = (_Float16*)(ws + qkv_bytes + obuf_bytes);
    _Float16* WpT  = (_Float16*)(ws + qkv_bytes + obuf_bytes + wat_bytes);

    // One-time weight transpose + f16 convert (streams through L2 once).
    transpose_w_kernel<<<dim3(N3c / 32, Ec / 32), dim3(32, 8), 0, stream>>>(Wa, WaT, Ec, N3c);
    transpose_w_kernel<<<dim3(Ec / 32, Ec / 32), dim3(32, 8), 0, stream>>>(Wp, WpT, Ec, Ec);

    qkv_rope_kernel<<<dim3(N3c / 64, (Bc * Sc) / 32), 32, 0, stream>>>(hs, WaT, ba, qkv);
    flash_attn_kernel<<<dim3(Sc / 16, Bc * Hc), 32, 0, stream>>>(qkv, obuf);
    proj_kernel<<<dim3(Ec / 64, (Bc * Sc) / 32), 32, 0, stream>>>(obuf, WpT, bp, out);
}
